// NodeEmbeddingLayer_19559281066592
// MI455X (gfx1250) — compile-verified, matrix operand order checked by reading, not run on hardware
//
#include <hip/hip_runtime.h>
#include <hip/hip_bf16.h>

// ---------------------------------------------------------------------------
// NodeEmbeddingLayer fused pipeline for gfx1250 (MI455X), wave32 + WMMA f16.
//
//   cw[n,f]  = (1/16) sum_k attn[n,k]*ctx[n,k,f]          (bandwidth kernel)
//   A1       = [x | cw]  (f16, [N,512])
//   u_pre    = A1 @ [Wn|Wc].T + bn + bc*mean(attn)        (WMMA GEMM, K=512)
//   u        = u_pre @ Wu.T + bu                          (WMMA GEMM, K=512)
//   ef[n,i*6+c] = c<5 ? Bspline_c(u[n,i]) : silu(u[n,i])  (fused in GEMM2 epilogue)
//   out      = ef @ [spline_w*scaler | base_w].T          (WMMA GEMM, K=1536)
// ---------------------------------------------------------------------------

typedef __attribute__((ext_vector_type(16))) _Float16     v16h;
typedef __attribute__((ext_vector_type(8)))  float        v8f;
typedef __attribute__((ext_vector_type(4)))  unsigned int u32x4;

#define NN 20000
#define FF 256
#define HH 512
#define OO 256
#define KD1 512
#define KD3 1536   // 256 * 6 augmented KAN features

union FragU { v16h v; u32x4 q[2]; };

__device__ __forceinline__ v8f zero8() {
  v8f z;
#pragma unroll
  for (int i = 0; i < 8; ++i) z[i] = 0.0f;
  return z;
}

// A-fragment layout per ISA 7.12.2 (16-bit A 16x32): lane l holds row (l&15);
// halves 0..7  = K [kb .. kb+7], halves 8..15 = K [kb+16 .. kb+23], kb=(l>=16)?8:0.
// B (32x16) is loaded from the transposed weight matrix Bt[N][K] with the
// mirrored layout: lane l holds col (l&15), same K striping. Both reduce to
// two 16-byte contiguous loads per lane -> global_load_b128.
__device__ __forceinline__ v16h load_frag_f16(const _Float16* __restrict__ p,
                                              int ld, int row, int k0, int lane) {
  const _Float16* base = p + (size_t)row * ld + k0 + ((lane >> 4) << 3);
  FragU u;
  u.q[0] = *(const u32x4*)(base);
  u.q[1] = *(const u32x4*)(base + 16);
  return u.v;
}

struct Acc24 { v8f a[2][4]; };

// 256-thread block = 8 waves arranged 4 (rows) x 2 (cols).
// Wave tile: 32 rows x 64 cols -> block tile 128 x 128.
__device__ __forceinline__ void gemm_core(const _Float16* __restrict__ A,
                                          const _Float16* __restrict__ Bt,
                                          int M, int Kd, Acc24& acc,
                                          int& m0, int& n0) {
  const int lane = threadIdx.x & 31;
  const int wid  = threadIdx.x >> 5;
  const int wr   = wid & 3;
  const int wc   = wid >> 2;
  m0 = blockIdx.x * 128 + wr * 32;
  n0 = blockIdx.y * 128 + wc * 64;
#pragma unroll
  for (int r = 0; r < 2; ++r)
#pragma unroll
    for (int c = 0; c < 4; ++c) acc.a[r][c] = zero8();

  for (int k0 = 0; k0 < Kd; k0 += 32) {
    v16h afr[2];
#pragma unroll
    for (int r = 0; r < 2; ++r) {
      int row = m0 + r * 16 + (lane & 15);
      if (row >= M) row = M - 1;   // branchless clamp; extra rows discarded at store
      afr[r] = load_frag_f16(A, Kd, row, k0, lane);
    }
#pragma unroll
    for (int c = 0; c < 4; ++c) {
      v16h bfr = load_frag_f16(Bt, Kd, n0 + c * 16 + (lane & 15), k0, lane);
#pragma unroll
      for (int r = 0; r < 2; ++r) {
        acc.a[r][c] = __builtin_amdgcn_wmma_f32_16x16x32_f16(
            false, afr[r], false, bfr, (short)0, acc.a[r][c], false, false);
      }
    }
  }
}

// C/D layout per ISA: lane l, vgpr j -> row m0 + ((l>=16)?8:0) + j, col n0 + (l&15).
#define C_ROW(m0, r, lane, j) ((m0) + (r) * 16 + (((lane) >> 4) << 3) + (j))
#define C_COL(n0, c, lane)    ((n0) + (c) * 16 + ((lane) & 15))

// ---------------------------------------------------------------------------
// Quadratic B-spline basis (grid_size=3, order=2, uniform knots h=2/3,
// t_j = -7/3 + j*h) + silu, producing the 6 augmented KAN features.
// ---------------------------------------------------------------------------
__device__ __forceinline__ void kan_features(float u, _Float16 out[6]) {
  const float h    = 2.0f / 3.0f;
  const float rh   = 1.5f;        // 1/h
  const float r2h  = 0.75f;       // 1/(2h)
  const float t0   = -7.0f / 3.0f;
  float b0[7];
#pragma unroll
  for (int j = 0; j < 7; ++j) {
    float tj = t0 + j * h;
    b0[j] = (u >= tj && u < tj + h) ? 1.0f : 0.0f;
  }
  float b1[6];
#pragma unroll
  for (int j = 0; j < 6; ++j) {
    float tj = t0 + j * h;
    b1[j] = (u - tj) * rh * b0[j] + (tj + 2.0f * h - u) * rh * b0[j + 1];
  }
#pragma unroll
  for (int j = 0; j < 5; ++j) {
    float tj = t0 + j * h;
    out[j] = (_Float16)((u - tj) * r2h * b1[j] +
                        (tj + 3.0f * h - u) * r2h * b1[j + 1]);
  }
  out[5] = (_Float16)(u / (1.0f + __expf(-u)));   // silu
}

// ---------------------------------------------------------------------------
// Kernel 1: pack / convert all weights to f16 (also fuses spline_w*scaler and
// interleaves the augmented KAN weight as [o][i*6+c], c=5 -> base_w).
// ---------------------------------------------------------------------------
__global__ __launch_bounds__(256) void pack_weights_kernel(
    const float* __restrict__ Wn, const float* __restrict__ Wc,
    const float* __restrict__ Wu, const float* __restrict__ base_w,
    const float* __restrict__ spline_w, const float* __restrict__ scaler,
    _Float16* __restrict__ W1h, _Float16* __restrict__ Wuh,
    _Float16* __restrict__ Waugh) {
  int idx = blockIdx.x * 256 + threadIdx.x;
  const int S1 = HH * (2 * FF);          // 512*512
  const int S2 = OO * HH;                // 256*512
  const int S3 = OO * KD3;               // 256*1536
  if (idx < S1) {
    int hh = idx >> 9, k = idx & 511;
    float v = (k < FF) ? Wn[hh * FF + k] : Wc[hh * FF + (k - FF)];
    W1h[idx] = (_Float16)v;
  } else if (idx < S1 + S2) {
    int i2 = idx - S1;
    Wuh[i2] = (_Float16)Wu[i2];
  } else if (idx < S1 + S2 + S3) {
    int i3 = idx - (S1 + S2);
    int o = i3 / KD3, r = i3 % KD3, i = r / 6, c = r % 6;
    float v = (c < 5) ? spline_w[(o * OO + i) * 5 + c] * scaler[o * OO + i]
                      : base_w[o * OO + i];
    Waugh[i3] = (_Float16)v;
  }
}

// ---------------------------------------------------------------------------
// Kernel 2: attention-weighted context mean + build A1 = [x | cw] in f16.
// One block per node; fully coalesced reads of the 327MB contexts tensor.
// ---------------------------------------------------------------------------
__global__ __launch_bounds__(256) void ctx_reduce_kernel(
    const float* __restrict__ x, const float* __restrict__ ctx,
    const float* __restrict__ attn, _Float16* __restrict__ A1,
    float* __restrict__ amean) {
  const int n = blockIdx.x;
  const int f = threadIdx.x;
  __shared__ float sa[16];
  if (f < 16) sa[f] = attn[n * 16 + f];
  __syncthreads();
  const float* cp = ctx + (size_t)n * 16 * FF + f;
  float acc = 0.0f;
#pragma unroll
  for (int k = 0; k < 16; ++k) acc += sa[k] * cp[(size_t)k * FF];
  A1[(size_t)n * (2 * FF) + f]      = (_Float16)x[(size_t)n * FF + f];
  A1[(size_t)n * (2 * FF) + FF + f] = (_Float16)(acc * (1.0f / 16.0f));
  if (f == 0) {
    float s = 0.0f;
#pragma unroll
    for (int k = 0; k < 16; ++k) s += sa[k];
    amean[n] = s * (1.0f / 16.0f);
  }
}

// ---------------------------------------------------------------------------
// Kernel 3: u_pre[N,512] = A1 @ W1.T + bn + bc*amean   (f16 out)
// ---------------------------------------------------------------------------
__global__ __launch_bounds__(256) void gemm1_kernel(
    const _Float16* __restrict__ A1, const _Float16* __restrict__ W1h,
    const float* __restrict__ bn, const float* __restrict__ bc,
    const float* __restrict__ amean, _Float16* __restrict__ u_pre) {
  const int lane = threadIdx.x & 31;
  Acc24 acc; int m0, n0;
  gemm_core(A1, W1h, NN, KD1, acc, m0, n0);
#pragma unroll
  for (int r = 0; r < 2; ++r)
#pragma unroll
    for (int c = 0; c < 4; ++c) {
      int col = C_COL(n0, c, lane);
      float b0 = bn[col], b1 = bc[col];
#pragma unroll
      for (int j = 0; j < 8; ++j) {
        int row = C_ROW(m0, r, lane, j);
        if (row < NN) {
          float v = acc.a[r][c][j] + b0 + b1 * amean[row];
          u_pre[(size_t)row * HH + col] = (_Float16)v;
        }
      }
    }
}

// ---------------------------------------------------------------------------
// Kernel 4: u = u_pre @ Wu.T + bu, then fused KAN featureization ->
// ef[n, col*6 + (0..4)=spline bases, 5=silu] as f16 (12B per element, 3x u32).
// ---------------------------------------------------------------------------
__global__ __launch_bounds__(256) void gemm2_kan_kernel(
    const _Float16* __restrict__ u_pre, const _Float16* __restrict__ Wuh,
    const float* __restrict__ bu, _Float16* __restrict__ ef) {
  const int lane = threadIdx.x & 31;
  Acc24 acc; int m0, n0;
  gemm_core(u_pre, Wuh, NN, HH, acc, m0, n0);
  unsigned int* efw = (unsigned int*)ef;
#pragma unroll
  for (int r = 0; r < 2; ++r)
#pragma unroll
    for (int c = 0; c < 4; ++c) {
      int col = C_COL(n0, c, lane);
      float bb = bu[col];
#pragma unroll
      for (int j = 0; j < 8; ++j) {
        int row = C_ROW(m0, r, lane, j);
        if (row < NN) {
          float u = acc.a[r][c][j] + bb;
          union { _Float16 h[6]; unsigned int w[3]; } pk;
          kan_features(u, pk.h);
          size_t base = (size_t)row * (KD3 / 2) + (size_t)col * 3;
          efw[base + 0] = pk.w[0];
          efw[base + 1] = pk.w[1];
          efw[base + 2] = pk.w[2];
        }
      }
    }
}

// ---------------------------------------------------------------------------
// Kernel 5: out[N,256] = ef @ Waug.T   (f32 out -> d_out)
// ---------------------------------------------------------------------------
__global__ __launch_bounds__(256) void gemm3_kernel(
    const _Float16* __restrict__ ef, const _Float16* __restrict__ Waugh,
    float* __restrict__ out) {
  const int lane = threadIdx.x & 31;
  Acc24 acc; int m0, n0;
  gemm_core(ef, Waugh, NN, KD3, acc, m0, n0);
#pragma unroll
  for (int r = 0; r < 2; ++r)
#pragma unroll
    for (int c = 0; c < 4; ++c) {
      int col = C_COL(n0, c, lane);
#pragma unroll
      for (int j = 0; j < 8; ++j) {
        int row = C_ROW(m0, r, lane, j);
        if (row < NN) out[(size_t)row * OO + col] = acc.a[r][c][j];
      }
    }
}

extern "C" void kernel_launch(void* const* d_in, const int* in_sizes, int n_in,
                              void* d_out, int out_size, void* d_ws, size_t ws_size,
                              hipStream_t stream) {
  (void)in_sizes; (void)n_in; (void)out_size; (void)ws_size;
  const float* x    = (const float*)d_in[0];
  const float* ctx  = (const float*)d_in[1];
  const float* attn = (const float*)d_in[2];
  const float* Wn   = (const float*)d_in[3];
  const float* bn   = (const float*)d_in[4];
  const float* Wc   = (const float*)d_in[5];
  const float* bc   = (const float*)d_in[6];
  const float* Wu   = (const float*)d_in[7];
  const float* bu   = (const float*)d_in[8];
  const float* kbw  = (const float*)d_in[9];
  const float* ksw  = (const float*)d_in[10];
  const float* ksc  = (const float*)d_in[11];
  float* out = (float*)d_out;

  char* ws = (char*)d_ws;
  size_t off = 0;
  auto carve = [&](size_t bytes) {
    size_t o = off;
    off = (off + bytes + 255) & ~(size_t)255;
    return o;
  };
  _Float16* W1h   = (_Float16*)(ws + carve((size_t)HH * 2 * FF * 2));   // 512 KB
  _Float16* Wuh   = (_Float16*)(ws + carve((size_t)OO * HH * 2));       // 256 KB
  _Float16* Waugh = (_Float16*)(ws + carve((size_t)OO * KD3 * 2));      // 768 KB
  float*    amean = (float*)   (ws + carve((size_t)NN * 4));            //  80 KB
  _Float16* A1    = (_Float16*)(ws + carve((size_t)NN * 2 * FF * 2));   // 20.5 MB
  _Float16* u_pre = (_Float16*)(ws + carve((size_t)NN * HH * 2));       // 20.5 MB
  _Float16* ef    = (_Float16*)(ws + carve((size_t)NN * KD3 * 2));      // 61.4 MB

  // 1. weight packing (786432 elements / 256)
  pack_weights_kernel<<<3072, 256, 0, stream>>>(Wn, Wc, Wu, kbw, ksw, ksc,
                                                W1h, Wuh, Waugh);
  // 2. context reduce + A1 build
  ctx_reduce_kernel<<<NN, 256, 0, stream>>>(x, ctx, attn, A1, amean);

  const int gx = (NN + 127) / 128;  // 157 row-blocks
  // 3. GEMM1: [N,512] x [512,512]
  gemm1_kernel<<<dim3(gx, HH / 128), 256, 0, stream>>>(A1, W1h, bn, bc, amean, u_pre);
  // 4. GEMM2 + KAN featureization: [N,512] x [512,256] -> ef[N,1536]
  gemm2_kan_kernel<<<dim3(gx, OO / 128), 256, 0, stream>>>(u_pre, Wuh, bu, ef);
  // 5. GEMM3: [N,1536] x [1536,256] -> out
  gemm3_kernel<<<dim3(gx, OO / 128), 256, 0, stream>>>(ef, Waugh, out);
}
